// MyRNNCell_6055903887833
// MI455X (gfx1250) — matrix operand order repaired and checked by reading, not compiled, not run on hardware
//
#include <hip/hip_runtime.h>
#include <math.h>

// ---------------------------------------------------------------------------
// RNN forward: h = tanh(x_t @ Wxh + h @ Whh); y_t = h @ Why
// B=256 T=512 D=512 H=1024 O=512
// 16 persistent workgroups, each owning a 16-row batch slice (one WMMA M-tile)
// for all T steps; only workgroup barriers needed. bf16 WMMA
// (v_wmma_f32_16x16x32_bf16) with f32 accumulation; weights pre-packed into
// B-fragment order and re-read from L2 each step. An opaque (volatile-asm)
// zero OFFSET — not an opaque pointer — stops LICM from hoisting/spilling the
// loop-invariant fragments while preserving pointer provenance, so the
// backend emits global_load_b128 (LOADcnt only), not flat_load_b128.
// Inner GEMM loops are software-pipelined (prefetch kt+1 while WMMA kt runs).
// ---------------------------------------------------------------------------

typedef __bf16 bf16;
typedef __attribute__((ext_vector_type(16))) __bf16 v16bf;
typedef __attribute__((ext_vector_type(8)))  __bf16 v8bf;
typedef __attribute__((ext_vector_type(8)))  float  v8f;
typedef __attribute__((ext_vector_type(4)))  float  v4f;

#define B_ 256
#define T_ 512
#define D_ 512
#define H_ 1024
#define O_ 512

// LDS row strides padded by 8 bf16 (16B) so the row stride is not a multiple
// of the 256B bank span -> A-fragment column reads rotate across banks.
#define XSTR (D_ + 8)   // 520
#define HSTR (H_ + 8)   // 1032

// Opaque zero produced by a volatile asm each loop iteration: addresses that
// add it cannot be hoisted by LICM (which otherwise pre-loads all B fragments
// and spills ~4KB/lane to scratch), but the base pointer's provenance is
// untouched, so address-space inference still yields GLOBAL loads.
__device__ __forceinline__ int opaque_zero() {
  int z = 0;
  asm volatile("" : "+s"(z));
  return z;
}

// --- WMMA fragment helpers -------------------------------------------------

// A-matrix 16x32 bf16 layout (ISA 7.12.2): lane (m = lane&15, g = lane>>4)
// holds elements 0..7  = K in [kbase + g*8,      +8)
//       elements 8..15 = K in [kbase + 16 + g*8, +8)
// -> two contiguous 16-byte LDS loads per lane.
__device__ __forceinline__ v16bf load_a_frag(const bf16* buf, int stride,
                                             int lane, int kbase) {
  const int m = lane & 15;
  const int g = lane >> 4;
  const bf16* p = buf + m * stride + kbase + g * 8;
  v8bf lo = *(const v8bf*)(p);
  v8bf hi = *(const v8bf*)(p + 16);
  return __builtin_shufflevector(lo, hi, 0, 1, 2, 3, 4, 5, 6, 7,
                                         8, 9, 10, 11, 12, 13, 14, 15);
}

// B-matrix fragments pre-packed so each lane's 16 values are contiguous:
// one aligned 32-byte global load.
__device__ __forceinline__ v16bf load_b_frag(const bf16* pk, int frag, int lane) {
  return *(const v16bf*)(pk + ((size_t)frag * 32 + lane) * 16);
}

__device__ __forceinline__ v8f wmma_bf16(v16bf a, v16bf b, v8f c) {
  // (neg_a, A, neg_b, B, c_mod, C, reuse_a, reuse_b)
  return __builtin_amdgcn_wmma_f32_16x16x32_bf16(false, a, false, b,
                                                 (short)0, c, false, false);
}

// --- Weight pre-pack: fp32 row-major [K x N] -> bf16 B-fragment order -------
// B-matrix 32x16 bf16 layout: lane (n = lane&15, g = lane>>4) holds
// K = kt*32 + g*8 + j for j = 0..7, K = kt*32 + 16 + g*8 + (j-8) for j = 8..15.
__global__ void pack_w_bf16(const float* __restrict__ W,
                            unsigned short* __restrict__ Pv, int K, int N) {
  bf16* P = (bf16*)Pv;
  int idx = blockIdx.x * blockDim.x + threadIdx.x;
  int KT = K >> 5;
  int total = KT * (N >> 4) * 32;
  if (idx >= total) return;
  int lane = idx & 31;
  int frag = idx >> 5;
  int kt = frag % KT;
  int nt = frag / KT;
  int n  = nt * 16 + (lane & 15);
  int kb = kt * 32 + (lane >> 4) * 8;
#pragma unroll
  for (int j = 0; j < 16; ++j)
    P[(size_t)idx * 16 + j] = (bf16)W[(size_t)(kb + j + (j >> 3) * 8) * N + n];
}

// --- Persistent recurrent kernel -------------------------------------------
// grid = 16 workgroups (one per 16-row batch slice), 1024 threads = 32 waves.
// Wave w owns h-ntiles {2w, 2w+1} of 64 and y-ntile w of 32.
__global__ __launch_bounds__(1024) void rnn_persistent(
    const float* __restrict__ x,
    const float* __restrict__ h0,
    const unsigned short* __restrict__ wxh_pk,
    const unsigned short* __restrict__ whh_pk,
    const unsigned short* __restrict__ why_pk,
    float* __restrict__ out_y,
    float* __restrict__ out_h) {

  __shared__ bf16 hbuf[16 * HSTR];   // h_t (bf16), padded rows
  __shared__ bf16 xbuf[16 * XSTR];   // x_t slice (bf16), padded rows

  const int tid  = threadIdx.x;
  const int lane = tid & 31;
  const int wave = tid >> 5;
  const int m0   = blockIdx.x * 16;      // batch-row base of this slice

  const int nt0 = wave * 2;              // first of 2 h-tiles (H/16 = 64)
  const int ot0 = wave;                  // y-tile (O/16 = 32)

  // init h from h0 input
  for (int i = tid; i < 16 * H_; i += 1024) {
    int m = i >> 10, c = i & (H_ - 1);
    hbuf[m * HSTR + c] = (bf16)h0[(size_t)(m0 + m) * H_ + c];
  }
  __syncthreads();

  // x staging assignment: 64 threads per row, 8 floats per thread
  const int xrow = tid >> 6;
  const int xcol = (tid & 63) * 8;

  for (int t = 0; t < T_; ++t) {
    // per-iteration opaque offset (defeat LICM / scratch spilling while
    // keeping GLOBAL address-space provenance of the weight pointers)
    const int z = opaque_zero();
    const bf16* wxh = (const bf16*)wxh_pk + z;
    const bf16* whh = (const bf16*)whh_pk + z;
    const bf16* why = (const bf16*)why_pk + z;

    // ---- stage x_t[m0..m0+15, :] into LDS as bf16 (streaming: NT loads) ----
    {
      const float* xg = x + ((size_t)(m0 + xrow) * T_ + t) * D_ + xcol;
      v4f a = __builtin_nontemporal_load((const v4f*)xg + 0);
      v4f b = __builtin_nontemporal_load((const v4f*)xg + 1);
      bf16* xb = xbuf + xrow * XSTR + xcol;
      xb[0] = (bf16)a[0]; xb[1] = (bf16)a[1]; xb[2] = (bf16)a[2]; xb[3] = (bf16)a[3];
      xb[4] = (bf16)b[0]; xb[5] = (bf16)b[1]; xb[6] = (bf16)b[2]; xb[7] = (bf16)b[3];
    }
    __syncthreads();

    // ---- h_next = x_t @ Wxh + h_t @ Whh (f32 acc), pipelined ----
    v8f acc0 = {0.f, 0.f, 0.f, 0.f, 0.f, 0.f, 0.f, 0.f};
    v8f acc1 = acc0;
    {
      // x part: KT = 16
      v16bf a  = load_a_frag(xbuf, XSTR, lane, 0);
      v16bf b0 = load_b_frag(wxh, (nt0 + 0) * (D_ / 32), lane);
      v16bf b1 = load_b_frag(wxh, (nt0 + 1) * (D_ / 32), lane);
      for (int kt = 0; kt < (D_ / 32) - 1; ++kt) {
        v16bf an  = load_a_frag(xbuf, XSTR, lane, (kt + 1) * 32);
        v16bf b0n = load_b_frag(wxh, (nt0 + 0) * (D_ / 32) + kt + 1, lane);
        v16bf b1n = load_b_frag(wxh, (nt0 + 1) * (D_ / 32) + kt + 1, lane);
        acc0 = wmma_bf16(a, b0, acc0);
        acc1 = wmma_bf16(a, b1, acc1);
        a = an; b0 = b0n; b1 = b1n;
      }
      acc0 = wmma_bf16(a, b0, acc0);
      acc1 = wmma_bf16(a, b1, acc1);
    }
    {
      // h part: KT = 32
      v16bf a  = load_a_frag(hbuf, HSTR, lane, 0);
      v16bf b0 = load_b_frag(whh, (nt0 + 0) * (H_ / 32), lane);
      v16bf b1 = load_b_frag(whh, (nt0 + 1) * (H_ / 32), lane);
      for (int kt = 0; kt < (H_ / 32) - 1; ++kt) {
        v16bf an  = load_a_frag(hbuf, HSTR, lane, (kt + 1) * 32);
        v16bf b0n = load_b_frag(whh, (nt0 + 0) * (H_ / 32) + kt + 1, lane);
        v16bf b1n = load_b_frag(whh, (nt0 + 1) * (H_ / 32) + kt + 1, lane);
        acc0 = wmma_bf16(a, b0, acc0);
        acc1 = wmma_bf16(a, b1, acc1);
        a = an; b0 = b0n; b1 = b1n;
      }
      acc0 = wmma_bf16(a, b0, acc0);
      acc1 = wmma_bf16(a, b1, acc1);
    }
    __syncthreads();   // every wave finished reading h_t

    // ---- tanh, write h_{t+1} back into hbuf (C/D layout: M = r + 8g) ----
    {
      const int g   = lane >> 4;
      const int nc0 = (nt0 + 0) * 16 + (lane & 15);
      const int nc1 = (nt0 + 1) * 16 + (lane & 15);
#pragma unroll
      for (int r = 0; r < 8; ++r) {
        const int m = r + 8 * g;
        hbuf[m * HSTR + nc0] = (bf16)tanhf(acc0[r]);
        hbuf[m * HSTR + nc1] = (bf16)tanhf(acc1[r]);
      }
    }
    __syncthreads();   // hbuf now holds h_{t+1}

    // ---- y_t = h_{t+1} @ Why, pipelined ----
    v8f accy = {0.f, 0.f, 0.f, 0.f, 0.f, 0.f, 0.f, 0.f};
    {
      v16bf a = load_a_frag(hbuf, HSTR, lane, 0);
      v16bf b = load_b_frag(why, ot0 * (H_ / 32), lane);
      for (int kt = 0; kt < (H_ / 32) - 1; ++kt) {
        v16bf an = load_a_frag(hbuf, HSTR, lane, (kt + 1) * 32);
        v16bf bn = load_b_frag(why, ot0 * (H_ / 32) + kt + 1, lane);
        accy = wmma_bf16(a, b, accy);
        a = an; b = bn;
      }
      accy = wmma_bf16(a, b, accy);
    }
    {
      const int g  = lane >> 4;
      const int oc = ot0 * 16 + (lane & 15);
#pragma unroll
      for (int r = 0; r < 8; ++r) {
        const int m = r + 8 * g;
        __builtin_nontemporal_store(
            accy[r], out_y + ((size_t)(m0 + m) * T_ + t) * O_ + oc);
      }
    }
    // next-iter xbuf overwrite is safe: all xbuf reads finished before the
    // post-accumulate barrier, and the x stage is after this iteration's
    // last barrier for every wave.
  }

  __syncthreads();
  // final hidden state (bf16 -> f32)
  for (int i = tid; i < 16 * H_; i += 1024) {
    int m = i >> 10, c = i & (H_ - 1);
    out_h[(size_t)(m0 + m) * H_ + c] = (float)hbuf[m * HSTR + c];
  }
}

// ---------------------------------------------------------------------------

extern "C" void kernel_launch(void* const* d_in, const int* in_sizes, int n_in,
                              void* d_out, int out_size, void* d_ws, size_t ws_size,
                              hipStream_t stream) {
  const float* x    = (const float*)d_in[0];   // [B,T,D]
  const float* h0   = (const float*)d_in[1];   // [B,H]
  const float* w_xh = (const float*)d_in[2];   // [D,H]
  const float* w_hh = (const float*)d_in[3];   // [H,H]
  const float* w_hy = (const float*)d_in[4];   // [H,O]

  // workspace: packed bf16 weight fragments (4 MB total)
  unsigned short* ws     = (unsigned short*)d_ws;
  unsigned short* wxh_pk = ws;                                     // 512*1024
  unsigned short* whh_pk = ws + (size_t)D_ * H_;                   // 1024*1024
  unsigned short* why_pk = ws + (size_t)D_ * H_ + (size_t)H_ * H_; // 1024*512

  const int thr_xh = (D_ / 32) * (H_ / 16) * 32;   // 32768
  const int thr_hh = (H_ / 32) * (H_ / 16) * 32;   // 65536
  const int thr_hy = (H_ / 32) * (O_ / 16) * 32;   // 32768
  pack_w_bf16<<<(thr_xh + 255) / 256, 256, 0, stream>>>(w_xh, wxh_pk, D_, H_);
  pack_w_bf16<<<(thr_hh + 255) / 256, 256, 0, stream>>>(w_hh, whh_pk, H_, H_);
  pack_w_bf16<<<(thr_hy + 255) / 256, 256, 0, stream>>>(w_hy, why_pk, H_, O_);

  float* out_y = (float*)d_out;                    // [B,T,O]
  float* out_h = out_y + (size_t)B_ * T_ * O_;     // [B,H]

  rnn_persistent<<<dim3(B_ / 16), dim3(1024), 0, stream>>>(
      x, h0, wxh_pk, whh_pk, why_pk, out_y, out_h);
}
